// SwinBlock_69904887710670
// MI455X (gfx1250) — compile-verified, matrix operand order checked by reading, not run on hardware
//
#include <hip/hip_runtime.h>
#include <hip/hip_bf16.h>

typedef __attribute__((ext_vector_type(16))) _Float16 v16h;
typedef __attribute__((ext_vector_type(8)))  float    v8f;

#define DIMC   384
#define NHEADS 12
#define HD     32
#define NWIN   49      // tokens per window
#define NB     2048    // B * (56/7)^2 = 32*64 windows
#define MROWS  100352  // NB * NWIN == B*L
#define HIDD   1536
#define QSCALE 0.17677669529663687f  // 32^-0.5

__device__ __forceinline__ v8f v8f_zero() {
    v8f z = {0.f,0.f,0.f,0.f,0.f,0.f,0.f,0.f};
    return z;
}

// ---------------------------------------------------------------- weight cvt
__global__ void f32_to_f16_kernel(const float* __restrict__ src,
                                  _Float16* __restrict__ dst, size_t n) {
    size_t i = (size_t)blockIdx.x * blockDim.x + threadIdx.x;
    if (i < n) dst[i] = (_Float16)src[i];
}

// ---------------------------------------------------------------- layernorm
// permute=1: fuse cyclic shift (-3,-3) + window partition into output row.
__global__ __launch_bounds__(128)
void ln_kernel(const float* __restrict__ x, const float* __restrict__ g,
               const float* __restrict__ bb, _Float16* __restrict__ out,
               int permute) {
    __shared__ float red[128];
    int row = blockIdx.x, tid = threadIdx.x;
    const float* xr = x + (size_t)row * DIMC;
    float v0 = xr[tid], v1 = xr[tid + 128], v2 = xr[tid + 256];
    red[tid] = v0 + v1 + v2;
    __syncthreads();
    for (int o = 64; o > 0; o >>= 1) {
        if (tid < o) red[tid] += red[tid + o];
        __syncthreads();
    }
    float mu = red[0] * (1.f / 384.f);
    __syncthreads();
    float d0 = v0 - mu, d1 = v1 - mu, d2 = v2 - mu;
    red[tid] = d0 * d0 + d1 * d1 + d2 * d2;
    __syncthreads();
    for (int o = 64; o > 0; o >>= 1) {
        if (tid < o) red[tid] += red[tid + o];
        __syncthreads();
    }
    float rstd = rsqrtf(red[0] * (1.f / 384.f) + 1e-5f);
    size_t drow;
    if (permute) {
        int b = row / 3136, p = row % 3136;
        int h0 = p / 56, w0 = p % 56;
        int hs = (h0 + 53) % 56, ws2 = (w0 + 53) % 56;  // roll(-3)
        int wh = hs / 7, r2 = hs % 7, ww = ws2 / 7, c2 = ws2 % 7;
        size_t win = ((size_t)b * 8 + wh) * 8 + ww;
        drow = win * NWIN + (r2 * 7 + c2);
    } else {
        drow = (size_t)row;
    }
    _Float16* o2 = out + drow * DIMC;
    o2[tid]       = (_Float16)(d0 * rstd * g[tid]       + bb[tid]);
    o2[tid + 128] = (_Float16)(d1 * rstd * g[tid + 128] + bb[tid + 128]);
    o2[tid + 256] = (_Float16)(d2 * rstd * g[tid + 256] + bb[tid + 256]);
}

// ---------------------------------------------------------------- GEMM core
// Wave computes a 32(M) x 64(N) tile of  A[M,K] @ W[Ncols,K]^T.
// A fragment: lane l holds row (base + l&15), K halfs [(l>>4)*16 .. +15].
// B fragment: lane l holds W row (col base + l&15), same K slice.
__device__ __forceinline__ void wave_gemm_32x64(const _Float16* __restrict__ A,
                                                const _Float16* __restrict__ W,
                                                int K, size_t rowBase,
                                                int colBase, v8f c[2][4]) {
    int l = threadIdx.x & 31;
    int lr = l & 15, lh = l >> 4;
    const _Float16* a0p = A + (rowBase + lr) * (size_t)K + lh * 16;
    const _Float16* a1p = a0p + (size_t)16 * K;
    const _Float16* bp  = W + ((size_t)colBase + lr) * (size_t)K + lh * 16;
    for (int kk = 0; kk < K; kk += 32) {
        v16h a0 = *(const v16h*)(a0p + kk);
        v16h a1 = *(const v16h*)(a1p + kk);
#pragma unroll
        for (int nt = 0; nt < 4; ++nt) {
            v16h b = *(const v16h*)(bp + (size_t)nt * 16 * K + kk);
            c[0][nt] = __builtin_amdgcn_wmma_f32_16x16x32_f16(
                false, a0, false, b, (short)0, c[0][nt], false, false);
            c[1][nt] = __builtin_amdgcn_wmma_f32_16x16x32_f16(
                false, a1, false, b, (short)0, c[1][nt], false, false);
        }
    }
}

// ---------------------------------------------------------------- QKV GEMM
// out layout: qkv[(((win*3 + t)*12 + head)*49 + n)*32 + d], q pre-scaled.
__global__ __launch_bounds__(256)
void gemm_qkv(const _Float16* __restrict__ A, const _Float16* __restrict__ W,
              const float* __restrict__ bias, _Float16* __restrict__ qkv) {
    int wv = threadIdx.x >> 5;
    size_t rowBase = ((size_t)blockIdx.y * 8 + wv) * 32;
    int colBase = blockIdx.x * 64;
    v8f c[2][4];
#pragma unroll
    for (int i = 0; i < 2; ++i)
#pragma unroll
        for (int j = 0; j < 4; ++j) c[i][j] = v8f_zero();
    wave_gemm_32x64(A, W, DIMC, rowBase, colBase, c);
    int l = threadIdx.x & 31, lr = l & 15, lh = l >> 4;
#pragma unroll
    for (int mi = 0; mi < 2; ++mi)
#pragma unroll
        for (int nt = 0; nt < 4; ++nt)
#pragma unroll
            for (int j = 0; j < 8; ++j) {
                size_t m = rowBase + mi * 16 + j + lh * 8;
                int cc = colBase + nt * 16 + lr;
                float val = c[mi][nt][j] + bias[cc];
                int t = cc / DIMC, rem = cc % DIMC;
                int head = rem >> 5, d = rem & 31;
                if (t == 0) val *= QSCALE;
                size_t win = m / NWIN;
                int n = (int)(m % NWIN);
                qkv[((((win * 3 + t) * NHEADS + head) * NWIN) + n) * HD + d] =
                    (_Float16)val;
            }
}

// ---------------------------------------------------------------- attention
// one block = one (window, head); 4 waves, 128 threads.
__global__ __launch_bounds__(128)
void attn_kernel(const _Float16* __restrict__ qkv,
                 const float* __restrict__ rel_bias,
                 const float* __restrict__ mask, _Float16* __restrict__ o_win) {
    int win = blockIdx.x / NHEADS, head = blockIdx.x % NHEADS;
    int tid = threadIdx.x;
    __shared__ _Float16 sq[64 * 32];
    __shared__ _Float16 sk[64 * 32];
    __shared__ _Float16 svt[32 * 64];   // v transposed: [d][k]
    __shared__ float    sS[64 * 64];
    __shared__ _Float16 sP[64 * 64];

    const _Float16* qg = qkv + (((size_t)win * 3 + 0) * NHEADS + head) *
                                   (size_t)(NWIN * HD);
    const _Float16* kg = qg + (size_t)NHEADS * NWIN * HD;
    const _Float16* vg = qg + (size_t)2 * NHEADS * NWIN * HD;

    // stage q,k row-major and v transposed, 128-bit global loads.
    // 64 rows x 4 segments of 8 halfs = 256 chunks; rows >= 49 are zero pad.
    for (int chunk = tid; chunk < 256; chunk += 128) {
        int r = chunk >> 2, seg = chunk & 3;
        uint4 zq = {0u, 0u, 0u, 0u}, zk = zq, zv = zq;
        if (r < NWIN) {
            zq = *(const uint4*)(qg + r * HD + seg * 8);
            zk = *(const uint4*)(kg + r * HD + seg * 8);
            zv = *(const uint4*)(vg + r * HD + seg * 8);
        }
        *(uint4*)(sq + r * HD + seg * 8) = zq;
        *(uint4*)(sk + r * HD + seg * 8) = zk;
        const _Float16* pv = (const _Float16*)&zv;
        int d0 = seg * 8;
#pragma unroll
        for (int i = 0; i < 8; ++i) svt[(d0 + i) * 64 + r] = pv[i];
    }
    __syncthreads();

    int wv = tid >> 5, l = tid & 31, lr = l & 15, lh = l >> 4;

    // S = q @ k^T  (+bias +mask), padded cols/rows -> -inf
    {
        v16h a = *(const v16h*)(sq + (wv * 16 + lr) * 32 + lh * 16);
#pragma unroll
        for (int nt = 0; nt < 4; ++nt) {
            v16h b = *(const v16h*)(sk + (nt * 16 + lr) * 32 + lh * 16);
            v8f cfr = v8f_zero();
            cfr = __builtin_amdgcn_wmma_f32_16x16x32_f16(
                false, a, false, b, (short)0, cfr, false, false);
#pragma unroll
            for (int j = 0; j < 8; ++j) {
                int r = wv * 16 + j + lh * 8;
                int cc = nt * 16 + lr;
                float val;
                if (r < NWIN && cc < NWIN) {
                    int hi = r / 7, wi = r % 7, hj = cc / 7, wj = cc % 7;
                    int ridx = (hi - hj + 6) * 13 + (wi - wj + 6);
                    val = cfr[j] + rel_bias[ridx * NHEADS + head] +
                          mask[((size_t)(win & 63) * NWIN + r) * NWIN + cc];
                } else {
                    val = -1e30f;
                }
                sS[r * 64 + cc] = val;
            }
        }
    }
    __syncthreads();

    // softmax rows 0..48
    if (tid < NWIN) {
        float mx = -1e30f;
        for (int j = 0; j < NWIN; ++j) mx = fmaxf(mx, sS[tid * 64 + j]);
        float sum = 0.f;
        for (int j = 0; j < NWIN; ++j) {
            float e = __expf(sS[tid * 64 + j] - mx);
            sum += e;
            sS[tid * 64 + j] = e;
        }
        float inv = 1.f / sum;
        for (int j = 0; j < NWIN; ++j)
            sP[tid * 64 + j] = (_Float16)(sS[tid * 64 + j] * inv);
        for (int j = NWIN; j < 64; ++j) sP[tid * 64 + j] = (_Float16)0.f;
    }
    for (int idx = tid; idx < (64 - NWIN) * 64; idx += 128)
        sP[NWIN * 64 + idx] = (_Float16)0.f;  // zero pad rows
    __syncthreads();

    // O = P @ v   (K=64 padded, N=32)
#pragma unroll
    for (int nt = 0; nt < 2; ++nt) {
        v8f cfr = v8f_zero();
#pragma unroll
        for (int ks = 0; ks < 2; ++ks) {
            v16h a = *(const v16h*)(sP + (wv * 16 + lr) * 64 + ks * 32 + lh * 16);
            v16h b = *(const v16h*)(svt + (nt * 16 + lr) * 64 + ks * 32 + lh * 16);
            cfr = __builtin_amdgcn_wmma_f32_16x16x32_f16(
                false, a, false, b, (short)0, cfr, false, false);
        }
#pragma unroll
        for (int j = 0; j < 8; ++j) {
            int r = wv * 16 + j + lh * 8;
            int d = nt * 16 + lr;
            if (r < NWIN)
                o_win[((size_t)win * NWIN + r) * DIMC + head * HD + d] =
                    (_Float16)cfr[j];
        }
    }
}

// ---------------------------------------------------------------- proj GEMM
// fuse window-reverse + un-shift + residual -> x1 (f32, image layout)
__global__ __launch_bounds__(256)
void gemm_proj(const _Float16* __restrict__ A, const _Float16* __restrict__ W,
               const float* __restrict__ bias, const float* __restrict__ xin,
               float* __restrict__ x1) {
    int wv = threadIdx.x >> 5;
    size_t rowBase = ((size_t)blockIdx.y * 8 + wv) * 32;
    int colBase = blockIdx.x * 64;
    v8f c[2][4];
#pragma unroll
    for (int i = 0; i < 2; ++i)
#pragma unroll
        for (int j = 0; j < 4; ++j) c[i][j] = v8f_zero();
    wave_gemm_32x64(A, W, DIMC, rowBase, colBase, c);
    int l = threadIdx.x & 31, lr = l & 15, lh = l >> 4;
#pragma unroll
    for (int mi = 0; mi < 2; ++mi)
#pragma unroll
        for (int nt = 0; nt < 4; ++nt)
#pragma unroll
            for (int j = 0; j < 8; ++j) {
                size_t m = rowBase + mi * 16 + j + lh * 8;
                int cc = colBase + nt * 16 + lr;
                float val = c[mi][nt][j] + bias[cc];
                size_t win = m / NWIN;
                int n = (int)(m % NWIN);
                int b = (int)(win >> 6), wh = (int)((win >> 3) & 7),
                    ww = (int)(win & 7);
                int r2 = n / 7, c2 = n % 7;
                int hs = wh * 7 + r2, ws2 = ww * 7 + c2;
                int h0 = (hs + 3) % 56, w0 = (ws2 + 3) % 56;  // reverse roll
                size_t dst = (((size_t)b * 3136) + h0 * 56 + w0) * DIMC + cc;
                x1[dst] = xin[dst] + val;
            }
}

// ---------------------------------------------------------------- fc1 GEMM
__device__ __forceinline__ float gelu_tanh(float x) {
    float x3 = x * x * x;
    float t = tanhf(0.7978845608028654f * (x + 0.044715f * x3));
    return 0.5f * x * (1.f + t);
}

__global__ __launch_bounds__(256)
void gemm_fc1(const _Float16* __restrict__ A, const _Float16* __restrict__ W,
              const float* __restrict__ bias, _Float16* __restrict__ hid) {
    int wv = threadIdx.x >> 5;
    size_t rowBase = ((size_t)blockIdx.y * 8 + wv) * 32;
    int colBase = blockIdx.x * 64;
    v8f c[2][4];
#pragma unroll
    for (int i = 0; i < 2; ++i)
#pragma unroll
        for (int j = 0; j < 4; ++j) c[i][j] = v8f_zero();
    wave_gemm_32x64(A, W, DIMC, rowBase, colBase, c);
    int l = threadIdx.x & 31, lr = l & 15, lh = l >> 4;
#pragma unroll
    for (int mi = 0; mi < 2; ++mi)
#pragma unroll
        for (int nt = 0; nt < 4; ++nt)
#pragma unroll
            for (int j = 0; j < 8; ++j) {
                size_t m = rowBase + mi * 16 + j + lh * 8;
                int cc = colBase + nt * 16 + lr;
                float val = gelu_tanh(c[mi][nt][j] + bias[cc]);
                hid[m * HIDD + cc] = (_Float16)val;
            }
}

// ---------------------------------------------------------------- fc2 GEMM
__global__ __launch_bounds__(256)
void gemm_fc2(const _Float16* __restrict__ A, const _Float16* __restrict__ W,
              const float* __restrict__ bias, const float* __restrict__ x1,
              float* __restrict__ out) {
    int wv = threadIdx.x >> 5;
    size_t rowBase = ((size_t)blockIdx.y * 8 + wv) * 32;
    int colBase = blockIdx.x * 64;
    v8f c[2][4];
#pragma unroll
    for (int i = 0; i < 2; ++i)
#pragma unroll
        for (int j = 0; j < 4; ++j) c[i][j] = v8f_zero();
    wave_gemm_32x64(A, W, HIDD, rowBase, colBase, c);
    int l = threadIdx.x & 31, lr = l & 15, lh = l >> 4;
#pragma unroll
    for (int mi = 0; mi < 2; ++mi)
#pragma unroll
        for (int nt = 0; nt < 4; ++nt)
#pragma unroll
            for (int j = 0; j < 8; ++j) {
                size_t m = rowBase + mi * 16 + j + lh * 8;
                int cc = colBase + nt * 16 + lr;
                float val = c[mi][nt][j] + bias[cc];
                size_t idx = m * DIMC + cc;
                out[idx] = x1[idx] + val;
            }
}

// ---------------------------------------------------------------- launch
extern "C" void kernel_launch(void* const* d_in, const int* in_sizes, int n_in,
                              void* d_out, int out_size, void* d_ws,
                              size_t ws_size, hipStream_t stream) {
    const float* x       = (const float*)d_in[0];
    const float* mask    = (const float*)d_in[1];
    const float* w_ln1   = (const float*)d_in[2];
    const float* b_ln1   = (const float*)d_in[3];
    const float* w_qkv   = (const float*)d_in[4];
    const float* b_qkv   = (const float*)d_in[5];
    const float* relbias = (const float*)d_in[6];
    const float* w_proj  = (const float*)d_in[7];
    const float* b_proj  = (const float*)d_in[8];
    const float* w_ln2   = (const float*)d_in[9];
    const float* b_ln2   = (const float*)d_in[10];
    const float* w_fc1   = (const float*)d_in[11];
    const float* b_fc1   = (const float*)d_in[12];
    const float* w_fc2   = (const float*)d_in[13];
    const float* b_fc2   = (const float*)d_in[14];
    float* out = (float*)d_out;

    char* ws = (char*)d_ws;
    size_t off = 0;
    auto alloc = [&](size_t bytes) -> void* {
        void* p = ws + off;
        off = (off + bytes + 255) & ~(size_t)255;
        return p;
    };
    _Float16* wqkv_h  = (_Float16*)alloc((size_t)3 * DIMC * DIMC * 2);
    _Float16* wproj_h = (_Float16*)alloc((size_t)DIMC * DIMC * 2);
    _Float16* wfc1_h  = (_Float16*)alloc((size_t)HIDD * DIMC * 2);
    _Float16* wfc2_h  = (_Float16*)alloc((size_t)DIMC * HIDD * 2);
    // region A (77MB): h_win -> o_win -> h2 (sequentially live)
    _Float16* regA = (_Float16*)alloc((size_t)MROWS * DIMC * 2);
    // region B (308MB): qkv (231MB) then hidden (308MB)
    _Float16* regB = (_Float16*)alloc((size_t)MROWS * HIDD * 2);
    float* x1 = (float*)alloc((size_t)MROWS * DIMC * 4);

    _Float16* h_win = regA;
    _Float16* o_win = regA;
    _Float16* h2    = regA;
    _Float16* qkv   = regB;
    _Float16* hid   = regB;

    // 1) weight conversion
    {
        size_t n0 = (size_t)3 * DIMC * DIMC, n1 = (size_t)DIMC * DIMC,
               n2 = (size_t)HIDD * DIMC, n3 = (size_t)DIMC * HIDD;
        f32_to_f16_kernel<<<(unsigned)((n0 + 255) / 256), 256, 0, stream>>>(
            w_qkv, wqkv_h, n0);
        f32_to_f16_kernel<<<(unsigned)((n1 + 255) / 256), 256, 0, stream>>>(
            w_proj, wproj_h, n1);
        f32_to_f16_kernel<<<(unsigned)((n2 + 255) / 256), 256, 0, stream>>>(
            w_fc1, wfc1_h, n2);
        f32_to_f16_kernel<<<(unsigned)((n3 + 255) / 256), 256, 0, stream>>>(
            w_fc2, wfc2_h, n3);
    }
    // 2) LN1 + shift + window partition
    ln_kernel<<<MROWS, 128, 0, stream>>>(x, w_ln1, b_ln1, h_win, 1);
    // 3) QKV GEMM (M=100352, K=384, N=1152)
    gemm_qkv<<<dim3(1152 / 64, MROWS / 256), 256, 0, stream>>>(h_win, wqkv_h,
                                                               b_qkv, qkv);
    // 4) attention per (window, head)
    attn_kernel<<<NB * NHEADS, 128, 0, stream>>>(qkv, relbias, mask, o_win);
    // 5) proj GEMM + window reverse + residual
    gemm_proj<<<dim3(DIMC / 64, MROWS / 256), 256, 0, stream>>>(
        o_win, wproj_h, b_proj, x, x1);
    // 6) LN2
    ln_kernel<<<MROWS, 128, 0, stream>>>(x1, w_ln2, b_ln2, h2, 0);
    // 7) fc1 + GELU
    gemm_fc1<<<dim3(HIDD / 64, MROWS / 256), 256, 0, stream>>>(h2, wfc1_h,
                                                               b_fc1, hid);
    // 8) fc2 + residual -> out
    gemm_fc2<<<dim3(DIMC / 64, MROWS / 256), 256, 0, stream>>>(hid, wfc2_h,
                                                               b_fc2, x1, out);
}
